// LocAttention_19207093748540
// MI455X (gfx1250) — compile-verified, matrix operand order checked by reading, not run on hardware
//
#include <hip/hip_runtime.h>
#include <hip/hip_bf16.h>
#include <stdint.h>

// ---------------------------------------------------------------------------
// MI455X (gfx1250) fused location-aware attention.
//   N=32, T=2048, ENC=1024, ATT=512, C=10, K-taps=129
// v3: TDM A-tile staging (tensor_load_to_lds + s_wait_tensorcnt, double-
// buffered) feeding bf16 WMMA; 6-arg builtin form for this toolchain.
// enc_pad (256MB) read exactly twice -> ~22us HBM floor @ 23.3 TB/s.
// ---------------------------------------------------------------------------

#define NB   32
#define TT   2048
#define ENC  1024
#define ATT  512
#define CCH  10
#define KTAP 129
#define LCTX 64

typedef __attribute__((ext_vector_type(16))) __bf16        v16bf;
typedef __attribute__((ext_vector_type(8)))  float         v8f;
typedef __attribute__((ext_vector_type(4)))  unsigned int  v4u;
typedef __attribute__((ext_vector_type(8)))  int           v8i;
typedef __attribute__((ext_vector_type(4)))  int           v4i;

union frag_t { uint4 q[2]; v16bf v; };

__device__ __forceinline__ uint32_t pack_bf16(float lo, float hi) {
  uint32_t ul = __float_as_uint(lo), uh = __float_as_uint(hi);
  uint32_t rl = (ul + 0x7FFFu + ((ul >> 16) & 1u)) >> 16;
  uint32_t rh = (uh + 0x7FFFu + ((uh >> 16) & 1u)) >> 16;
  return (rh << 16) | (rl & 0xFFFFu);
}

// ISA 7.12.2: 16-bit 16x32 A/B fragment. Lane half (lane>>4) picks the K
// octet; VGPR i holds bf16 pair (k0, k0+1).
__device__ __forceinline__ int frag_k0(int lane, int i) {
  int half = lane >> 4;
  return (i < 4) ? (half * 8 + 2 * i) : (16 + half * 8 + 2 * (i - 4));
}

// TDM 2D tile load: 32 rows x 32 fp32, row pitch ENC floats. D# per ISA §8.
__device__ __forceinline__ void tdm_load_tile(const float* gptr, uint32_t lds_off) {
  uint64_t ga = (uint64_t)(uintptr_t)gptr;
  v4u g0;
  g0[0] = 1u;                                              // count=1, user mode
  g0[1] = lds_off;                                         // lds_addr [63:32]
  g0[2] = (uint32_t)ga;                                    // global_addr lo
  g0[3] = (uint32_t)((ga >> 32) & 0x01FFFFFFu) | (2u << 30); // ga[56:32]|type=2
  v8i g1;
  g1[0] = 0x00020000;            // wg_mask=0, data_size=2 (4B), no pad/iter
  g1[1] = (int)(32u << 16);      // tensor_dim0[15:0]=32 @bits63:48
  g1[2] = (int)(32u << 16);      // tensor_dim1[15:0]=32 @bits95:80
  g1[3] = (int)(32u << 16);      // tile_dim0=32        @bits127:112
  g1[4] = 32;                    // tile_dim1=32        @bits143:128
  g1[5] = ENC;                   // tensor_dim0_stride=1024 @bits191:160
  g1[6] = 0;
  g1[7] = 0;
  v4i z4 = {0, 0, 0, 0};
  v8i z8 = {0, 0, 0, 0, 0, 0, 0, 0};
  __builtin_amdgcn_tensor_load_to_lds(g0, g1, z4, z4, z8, 0);
}

// --- prep 1: dec_part[n][a] = dec_prev[n] . W_dec[:,a]  (32x512, trivial) ---
__global__ __launch_bounds__(256) void decpart_kernel(
    const float* __restrict__ dec_prev, const float* __restrict__ W_dec,
    float* __restrict__ dec_part) {
  int idx = blockIdx.x * 256 + threadIdx.x;   // 16384
  int n = idx >> 9, a = idx & 511;
  float s = 0.f;
  for (int d = 0; d < ENC; ++d)
    s += dec_prev[n * ENC + d] * W_dec[(size_t)d * ATT + a];
  dec_part[idx] = s;
}

// --- prep 2: g[n][t][c] = F_b[c] + sum_k ali_prev[n][t-64+k] * F_w[c][k] ---
__global__ __launch_bounds__(256) void conv_kernel(
    const float* __restrict__ ali, const float* __restrict__ F_w,
    const float* __restrict__ F_b, float* __restrict__ g) {
  __shared__ float xs[256 + 2 * LCTX];
  __shared__ float ws[CCH * KTAP];
  __shared__ float fb[CCH];
  int n = blockIdx.x >> 3;            // 8 chunks of 256 t per n
  int t0 = (blockIdx.x & 7) * 256;
  int tid = threadIdx.x;
  for (int i = tid; i < CCH * KTAP; i += 256) ws[i] = F_w[i];
  if (tid < CCH) fb[tid] = F_b[tid];
  for (int i = tid; i < 256 + 2 * LCTX; i += 256) {
    int t = t0 - LCTX + i;
    xs[i] = (t >= 0 && t < TT) ? ali[n * TT + t] : 0.f;
  }
  __syncthreads();
  float acc[CCH];
#pragma unroll
  for (int c = 0; c < CCH; ++c) acc[c] = fb[c];
  for (int k = 0; k < KTAP; ++k) {
    float x = xs[tid + k];
#pragma unroll
    for (int c = 0; c < CCH; ++c) acc[c] += x * ws[c * KTAP + k];
  }
  size_t base = ((size_t)(n * TT + t0 + tid)) * CCH;
#pragma unroll
  for (int c = 0; c < CCH; ++c) g[base + c] = acc[c];
}

// --- prep 3: pack W_enc (fp32 [K=1024][A=512]) into bf16 B-fragment layout --
__global__ __launch_bounds__(256) void pack_kernel(
    const float* __restrict__ W_enc, uint32_t* __restrict__ Wpack) {
  int idx = blockIdx.x * 256 + threadIdx.x;   // 262144 dwords (1 MB)
  int i = idx & 7;
  int lane = (idx >> 3) & 31;
  int kc = (idx >> 8) & 31;
  int a_tile = idx >> 13;
  int a = a_tile * 16 + (lane & 15);
  int k = 32 * kc + frag_k0(lane, i);
  Wpack[idx] = pack_bf16(W_enc[(size_t)k * ATT + a],
                         W_enc[(size_t)(k + 1) * ATT + a]);
}

// --- main: fused score GEMM + epilogue ------------------------------------
// Workgroup = 32 rows (one n, 32 consecutive t) x all 512 A-columns.
// 8 waves; wave w owns A cols [w*64, w*64+64). TDM double-buffers A tiles.
__global__ __launch_bounds__(256) void score_kernel(
    const float* __restrict__ enc, const float* __restrict__ dec_part,
    const float* __restrict__ g, const uint32_t* __restrict__ Wpack,
    const float* __restrict__ b_enc, const float* __restrict__ W_att,
    const float* __restrict__ w_vec, float* __restrict__ score) {
  __shared__ __align__(16) float ldsT[2][32 * 32];   // fp32 A tiles (TDM dest)
  __shared__ float gsh[32][CCH];
  __shared__ float sAcc[32];

  int tid = threadIdx.x;
  int wave = tid >> 5, lane = tid & 31, laneM = lane & 15, half = lane >> 4;
  int Mbase = blockIdx.x * 32;        // T divisible by 32 -> single n
  int n = Mbase / TT, tbase = Mbase % TT;
  const float* encTile = enc + (size_t)Mbase * ENC;

  for (int idx = tid; idx < 32 * CCH; idx += 256) {
    int row = idx / CCH, c = idx % CCH;
    gsh[row][c] = g[((size_t)(n * TT + tbase + row)) * CCH + c];
  }
  if (tid < 32) sAcc[tid] = 0.f;

  uint32_t lds0 = (uint32_t)(uintptr_t)&ldsT[0][0];
  uint32_t lds1 = (uint32_t)(uintptr_t)&ldsT[1][0];

  if (wave == 0) tdm_load_tile(encTile, lds0);       // prologue: tile kc=0

  v8f acc[2][4] = {};

  for (int kc = 0; kc < 32; ++kc) {
    int buf = kc & 1;
    __syncthreads();   // everyone done reading buf^1 from previous iteration
    if (wave == 0) {
      if (kc < 31) {
        tdm_load_tile(encTile + (kc + 1) * 32, buf ? lds0 : lds1);
        __builtin_amdgcn_s_wait_tensorcnt(1);   // oldest (tile kc) landed
      } else {
        __builtin_amdgcn_s_wait_tensorcnt(0);
      }
    }
    __syncthreads();   // tile kc visible to all waves

    // Assemble bf16 A fragments from fp32 tile (ISA 7.12.2 layout).
    frag_t aF[2];
#pragma unroll
    for (int mt = 0; mt < 2; ++mt) {
      const float* rp = &ldsT[buf][(mt * 16 + laneM) * 32 + half * 8];
      float4 x0 = *(const float4*)(rp);        // k octet lo
      float4 x1 = *(const float4*)(rp + 4);
      float4 x2 = *(const float4*)(rp + 16);   // k octet hi (+16)
      float4 x3 = *(const float4*)(rp + 20);
      aF[mt].q[0] = make_uint4(pack_bf16(x0.x, x0.y), pack_bf16(x0.z, x0.w),
                               pack_bf16(x1.x, x1.y), pack_bf16(x1.z, x1.w));
      aF[mt].q[1] = make_uint4(pack_bf16(x2.x, x2.y), pack_bf16(x2.z, x2.w),
                               pack_bf16(x3.x, x3.y), pack_bf16(x3.z, x3.w));
    }
#pragma unroll
    for (int at = 0; at < 4; ++at) {
      int a_tile = wave * 4 + at;
      const uint32_t* bp = Wpack + (((size_t)a_tile * 32 + kc) * 32 + lane) * 8;
      frag_t bF;
      bF.q[0] = *(const uint4*)bp;
      bF.q[1] = *(const uint4*)(bp + 4);
#pragma unroll
      for (int mt = 0; mt < 2; ++mt)
        acc[mt][at] = __builtin_amdgcn_wmma_f32_16x16x32_bf16(
            false, aF[mt].v, false, bF.v, (short)0, acc[mt][at], false, false);
    }
  }

  // Epilogue: + b_enc + dec_part + (g . W_att^T), tanh, dot w_vec, reduce.
  float dpb[4], wv[4], wa[4][CCH];
#pragma unroll
  for (int at = 0; at < 4; ++at) {
    int a = wave * 64 + at * 16 + laneM;
    dpb[at] = dec_part[n * ATT + a] + b_enc[a];
    wv[at] = w_vec[a];
#pragma unroll
    for (int c = 0; c < CCH; ++c) wa[at][c] = W_att[a * CCH + c];
  }
  int rowoff = half * 8;   // C-layout: VGPR r -> row r (+8 upper half)
#pragma unroll
  for (int mt = 0; mt < 2; ++mt) {
#pragma unroll
    for (int r = 0; r < 8; ++r) {
      int row = mt * 16 + rowoff + r;
      float s = 0.f;
#pragma unroll
      for (int at = 0; at < 4; ++at) {
        float attv = 0.f;
#pragma unroll
        for (int c = 0; c < CCH; ++c) attv += gsh[row][c] * wa[at][c];
        float val = acc[mt][at][r] + dpb[at] + attv;
        s += wv[at] * tanhf(val);
      }
      atomicAdd(&sAcc[row], s);   // ds_add_f32
    }
  }
  __syncthreads();
  if (tid < 32) score[Mbase + tid] = sAcc[tid];
}

// --- softmax over valid t (mask t >= enc_len -> 0) -------------------------
__global__ __launch_bounds__(256) void softmax_kernel(
    const float* __restrict__ score, const int* __restrict__ enc_len,
    float* __restrict__ ali) {
  __shared__ float red[256];
  int n = blockIdx.x, tid = threadIdx.x;
  int len = enc_len[n];
  float mx = -INFINITY;
  for (int t = tid; t < TT; t += 256)
    if (t < len) mx = fmaxf(mx, score[n * TT + t]);
  red[tid] = mx; __syncthreads();
  for (int s = 128; s > 0; s >>= 1) {
    if (tid < s) red[tid] = fmaxf(red[tid], red[tid + s]);
    __syncthreads();
  }
  mx = red[0]; __syncthreads();
  float ev[8], sum = 0.f;
#pragma unroll
  for (int j = 0; j < 8; ++j) {
    int t = tid + j * 256;
    float e = (t < len) ? __expf(score[n * TT + t] - mx) : 0.f;
    ev[j] = e; sum += e;
  }
  red[tid] = sum; __syncthreads();
  for (int s = 128; s > 0; s >>= 1) {
    if (tid < s) red[tid] += red[tid + s];
    __syncthreads();
  }
  float inv = 1.f / red[0];
#pragma unroll
  for (int j = 0; j < 8; ++j) ali[n * TT + tid + j * 256] = ev[j] * inv;
}

// --- ctx[n][d] = sum_t ali[n][t] * enc[n][t][d]  (memory bound, 256MB) -----
__global__ __launch_bounds__(256) void ctx_kernel(
    const float* __restrict__ ali, const float* __restrict__ enc,
    float* __restrict__ ctx) {
  int n = blockIdx.x >> 2;
  int d = (blockIdx.x & 3) * 256 + threadIdx.x;
  const float* ep = enc + (size_t)n * TT * ENC + d;
  const float* ap = ali + n * TT;
  float s0 = 0.f, s1 = 0.f, s2 = 0.f, s3 = 0.f;
  for (int t = 0; t < TT; t += 4) {
    s0 += ap[t + 0] * ep[(size_t)(t + 0) * ENC];
    s1 += ap[t + 1] * ep[(size_t)(t + 1) * ENC];
    s2 += ap[t + 2] * ep[(size_t)(t + 2) * ENC];
    s3 += ap[t + 3] * ep[(size_t)(t + 3) * ENC];
  }
  ctx[n * ENC + d] = (s0 + s1) + (s2 + s3);
}

extern "C" void kernel_launch(void* const* d_in, const int* in_sizes, int n_in,
                              void* d_out, int out_size, void* d_ws, size_t ws_size,
                              hipStream_t stream) {
  (void)in_sizes; (void)n_in; (void)out_size; (void)ws_size;
  const float* enc_pad  = (const float*)d_in[0];
  const int*   enc_len  = (const int*)d_in[1];
  const float* dec_prev = (const float*)d_in[2];
  const float* ali_prev = (const float*)d_in[3];
  const float* W_enc    = (const float*)d_in[4];
  const float* b_enc    = (const float*)d_in[5];
  const float* W_dec    = (const float*)d_in[6];
  const float* F_w      = (const float*)d_in[7];
  const float* F_b      = (const float*)d_in[8];
  const float* W_att    = (const float*)d_in[9];
  const float* w_vec    = (const float*)d_in[10];

  float* out     = (float*)d_out;
  float* ali_out = out;                    // N*T = 65536
  float* ctx_out = out + NB * TT;          // N*ENC = 32768

  float* ws       = (float*)d_ws;
  float* dec_part = ws;                              // 16384 f
  float* g        = dec_part + NB * ATT;             // 655360 f
  float* score    = g + (size_t)NB * TT * CCH;       // 65536 f
  uint32_t* Wpack = (uint32_t*)(score + NB * TT);    // 262144 dw (1 MB)

  decpart_kernel<<<64,   256, 0, stream>>>(dec_prev, W_dec, dec_part);
  conv_kernel   <<<NB*8, 256, 0, stream>>>(ali_prev, F_w, F_b, g);
  pack_kernel   <<<1024, 256, 0, stream>>>(W_enc, Wpack);
  score_kernel  <<<NB*TT/32, 256, 0, stream>>>(enc_pad, dec_part, g, Wpack,
                                               b_enc, W_att, w_vec, score);
  softmax_kernel<<<NB,   256, 0, stream>>>(score, enc_len, ali_out);
  ctx_kernel    <<<NB*4, 256, 0, stream>>>(ali_out, enc_pad, ctx_out);
}
